// MSDformer_46127948759287
// MI455X (gfx1250) — matrix operand order repaired and checked by reading, not compiled, not run on hardware
//
#include <hip/hip_runtime.h>

// ---------------------------------------------------------------------------
// MSDformer for MI455X (gfx1250): bf16 WMMA GEMMs + fp32 VALU for sampling/
// softmax. Other matrix stages use v_wmma_f32_16x16x32_bf16; attn x V is fp32 VALU.
// GEMM: 256 thr = 8 waves (4x2), each wave owns a 2x2 grid of 16x16 tiles
// -> 4 WMMA per wave per K-chunk (BM=128, BN=64, BK=32).
// ---------------------------------------------------------------------------

typedef __attribute__((ext_vector_type(16))) unsigned short v16u;
typedef __attribute__((ext_vector_type(8)))  unsigned short u16x8;
typedef __attribute__((ext_vector_type(4)))  unsigned short u16x4;
typedef __attribute__((ext_vector_type(16))) __bf16         v16bf;
typedef __attribute__((ext_vector_type(8)))  float          v8f;

#define CDIM   192
#define HWDIM  9216      // 96*96
#define BATCH  2
#define MTOT   18432     // BATCH*HWDIM
#define KDEF   1728      // CDIM*9
#define NHEAD  6
#define HD     32
#define NOFF   64        // offset-conv N padded 18 -> 64

__device__ __forceinline__ unsigned short f32_to_bf16(float f) {
  unsigned int u = __float_as_uint(f);
  u += 0x7FFFu + ((u >> 16) & 1u);           // round-to-nearest-even
  return (unsigned short)(u >> 16);
}

// ---------------------------------------------------------------------------
// Weight repack: out[s*Rpad + r] = bf16(in[r*S + s]), zero pad r in [R,Rpad)
// ---------------------------------------------------------------------------
__global__ void pack_transpose_pad(const float* __restrict__ in,
                                   unsigned short* __restrict__ out,
                                   int R, int S, int Rpad) {
  int idx = blockIdx.x * blockDim.x + threadIdx.x;
  if (idx >= S * Rpad) return;
  int r = idx % Rpad, s = idx / Rpad;
  float v = (r < R) ? in[(long)r * S + s] : 0.0f;
  out[idx] = f32_to_bf16(v);
}

__global__ void f32_to_bf16_copy(const float* __restrict__ in,
                                 unsigned short* __restrict__ out, long n) {
  long i = (long)blockIdx.x * blockDim.x + threadIdx.x;
  if (i < n) out[i] = f32_to_bf16(in[i]);
}

// ---------------------------------------------------------------------------
// im2col for the 3x3/pad1 offset conv.  A^T layout: A[k=ci*9+kk][m=b*HW+hw].
// ---------------------------------------------------------------------------
__global__ void im2col_kernel(const float* __restrict__ x,
                              unsigned short* __restrict__ A) {
  long idx = (long)blockIdx.x * blockDim.x + threadIdx.x;
  const long total = (long)KDEF * MTOT;
  if (idx >= total) return;
  int m = (int)(idx % MTOT);
  int k = (int)(idx / MTOT);
  int hw = m % HWDIM, b = m / HWDIM;
  int h = hw / 96, w = hw % 96;
  int ci = k / 9, kk = k % 9;
  int hh = h + kk / 3 - 1, ww = w + kk % 3 - 1;
  float v = 0.0f;
  if (hh >= 0 && hh < 96 && ww >= 0 && ww < 96)
    v = x[((long)b * CDIM + ci) * HWDIM + hh * 96 + ww];
  A[idx] = f32_to_bf16(v);
}

// ---------------------------------------------------------------------------
// Deformable bilinear sampler: builds A^T[k=c*9+n][m=b*HW+hw] (bf16).
// One thread per (b,n,hw); 4 taps/weights computed once, reused for 192 ch.
// ---------------------------------------------------------------------------
__global__ void deform_sample_kernel(const float* __restrict__ x,
                                     const float* __restrict__ off,
                                     const float* __restrict__ pbias,
                                     unsigned short* __restrict__ A) {
  int idx = blockIdx.x * blockDim.x + threadIdx.x;   // B*9*HW
  if (idx >= BATCH * 9 * HWDIM) return;
  int hw = idx % HWDIM;
  int n  = (idx / HWDIM) % 9;
  int b  = idx / (HWDIM * 9);
  int h = hw / 96, w = hw % 96;

  float px = (float)(h + 1) + (float)(n / 3 - 1)
           + off[((long)b * NOFF + n) * HWDIM + hw] + pbias[n];
  float py = (float)(w + 1) + (float)(n % 3 - 1)
           + off[((long)b * NOFF + 9 + n) * HWDIM + hw] + pbias[9 + n];

  const float HPm1 = 97.0f;                           // Hp-1, padded frame
  float fx = floorf(px), fy = floorf(py);
  float qx0 = fminf(fmaxf(fx,        0.f), HPm1);
  float qy0 = fminf(fmaxf(fy,        0.f), HPm1);
  float qx1 = fminf(fmaxf(fx + 1.f,  0.f), HPm1);
  float qy1 = fminf(fmaxf(fy + 1.f,  0.f), HPm1);
  float pxc = fminf(fmaxf(px, 0.f), HPm1);
  float pyc = fminf(fmaxf(py, 0.f), HPm1);
  float glt = (1.f + (qx0 - pxc)) * (1.f + (qy0 - pyc));
  float grb = (1.f - (qx1 - pxc)) * (1.f - (qy1 - pyc));
  float glb = (1.f + (qx0 - pxc)) * (1.f - (qy1 - pyc));
  float grt = (1.f - (qx1 - pxc)) * (1.f + (qy0 - pyc));
  int ix0 = (int)qx0, iy0 = (int)qy0, ix1 = (int)qx1, iy1 = (int)qy1;

  float g0 = glt, g1 = grb, g2 = glb, g3 = grt;
  int i0, i1, i2, i3;
  // padded -> unpadded index; zero weight outside the valid interior
  if (ix0 < 1 || ix0 > 96 || iy0 < 1 || iy0 > 96) { g0 = 0.f; i0 = 0; }
  else i0 = (ix0 - 1) * 96 + (iy0 - 1);
  if (ix1 < 1 || ix1 > 96 || iy1 < 1 || iy1 > 96) { g1 = 0.f; i1 = 0; }
  else i1 = (ix1 - 1) * 96 + (iy1 - 1);
  if (ix0 < 1 || ix0 > 96 || iy1 < 1 || iy1 > 96) { g2 = 0.f; i2 = 0; }
  else i2 = (ix0 - 1) * 96 + (iy1 - 1);
  if (ix1 < 1 || ix1 > 96 || iy0 < 1 || iy0 > 96) { g3 = 0.f; i3 = 0; }
  else i3 = (ix1 - 1) * 96 + (iy0 - 1);

  const float* xb = x + (long)b * CDIM * HWDIM;
  long mcol = (long)b * HWDIM + hw;
  for (int c = 0; c < CDIM; ++c) {
    const float* xc = xb + (long)c * HWDIM;
    float v = g0 * xc[i0] + g1 * xc[i1] + g2 * xc[i2] + g3 * xc[i3];
    A[(long)(c * 9 + n) * MTOT + mcol] = f32_to_bf16(v);
  }
}

// ---------------------------------------------------------------------------
// Generic bf16 WMMA GEMM, NCHW output (HW = HWDIM compile-time).
//   C[(m/HWDIM)*Cout + n][m%HWDIM] (+batch offset) = sum_k A[m][k] * B[k][n]
// A stored transposed: A[k][m], row stride lda (==M). B row-major [K][N].
// Block = 256 threads = 8 waves in 4x2; each wave: 2x2 tiles of 16x16.
// BM=128, BN=64, BK=32 -> 4 WMMA per wave per K-chunk.
// HWDIM % 128 == 0, so a whole block lies in one batch image: bb/hwbase are
// per-block constants (no per-element division in the epilogue).
// ---------------------------------------------------------------------------
__global__ __launch_bounds__(256) void gemm_bf16_nchw(
    const unsigned short* __restrict__ A, const unsigned short* __restrict__ B,
    float* __restrict__ C, int K, int lda, int ldb,
    long a_bstride, long c_bstride, int Cout) {
  __shared__ __align__(16) unsigned short As[128][40];  // [m][k]
  __shared__ __align__(16) unsigned short Bs[64][40];   // N-major (Bs[n][k])

  const int tid  = threadIdx.x;
  const int lane = tid & 31;
  const int wave = tid >> 5;
  const int wm = wave & 3, wn = wave >> 2;     // 4x2 wave grid
  const int mr = lane & 15, hi = lane >> 4;
  const int m0 = blockIdx.y * 128;
  const int n0 = blockIdx.x * 64;

  const unsigned short* Ab = A + (long)blockIdx.z * a_bstride;
  float* Cb = C + (long)blockIdx.z * c_bstride;

  v8f acc00 = {0.f,0.f,0.f,0.f,0.f,0.f,0.f,0.f};
  v8f acc01 = acc00, acc10 = acc00, acc11 = acc00;

  const int kk = tid >> 3;                     // 0..31 (k row for both tiles)
  const int mc = (tid & 7) << 4;               // 0..112 (A: 16 m each)
  const int nc = (tid & 7) << 3;               // 0..56  (B: 8 n each)

  for (int k0 = 0; k0 < K; k0 += 32) {
    __syncthreads();
    {   // A tile: 32 k-rows x 128 m; each thread 16 contiguous m (2x16B)
      u16x8 v0 = *(const u16x8*)(Ab + (long)(k0 + kk) * lda + m0 + mc);
      u16x8 v1 = *(const u16x8*)(Ab + (long)(k0 + kk) * lda + m0 + mc + 8);
      #pragma unroll
      for (int i = 0; i < 8; ++i) As[mc + i][kk]     = v0[i];
      #pragma unroll
      for (int i = 0; i < 8; ++i) As[mc + 8 + i][kk] = v1[i];
    }
    {   // B tile: 32 k-rows x 64 n; write transposed into Bs[n][k]
      u16x8 v = *(const u16x8*)(B + (long)(k0 + kk) * ldb + n0 + nc);
      #pragma unroll
      for (int i = 0; i < 8; ++i) Bs[nc + i][kk] = v[i];
    }
    __syncthreads();

    union { v16u u; v16bf b; } af0, af1, bf0, bf1;
    {   // A fragments: rows wm*32 + {0,16} + mr
      int r0 = wm * 32 + mr;
      u16x8 a0 = *(const u16x8*)(&As[r0][hi * 8]);
      u16x8 a1 = *(const u16x8*)(&As[r0][16 + hi * 8]);
      u16x8 a2 = *(const u16x8*)(&As[r0 + 16][hi * 8]);
      u16x8 a3 = *(const u16x8*)(&As[r0 + 16][16 + hi * 8]);
      #pragma unroll
      for (int i = 0; i < 8; ++i) {
        af0.u[i] = a0[i]; af0.u[8 + i] = a1[i];
        af1.u[i] = a2[i]; af1.u[8 + i] = a3[i];
      }
    }
    {   // B fragments: cols wn*32 + {0,16} + mr (Bs is N-major)
      int c0 = wn * 32 + mr;
      u16x8 b0 = *(const u16x8*)(&Bs[c0][hi * 8]);
      u16x8 b1 = *(const u16x8*)(&Bs[c0][16 + hi * 8]);
      u16x8 b2 = *(const u16x8*)(&Bs[c0 + 16][hi * 8]);
      u16x8 b3 = *(const u16x8*)(&Bs[c0 + 16][16 + hi * 8]);
      #pragma unroll
      for (int i = 0; i < 8; ++i) {
        bf0.u[i] = b0[i]; bf0.u[8 + i] = b1[i];
        bf1.u[i] = b2[i]; bf1.u[8 + i] = b3[i];
      }
    }
    acc00 = __builtin_amdgcn_wmma_f32_16x16x32_bf16(false, af0.b, false, bf0.b, (short)0, acc00, false, false);
    acc01 = __builtin_amdgcn_wmma_f32_16x16x32_bf16(false, af0.b, false, bf1.b, (short)0, acc01, false, false);
    acc10 = __builtin_amdgcn_wmma_f32_16x16x32_bf16(false, af1.b, false, bf0.b, (short)0, acc10, false, false);
    acc11 = __builtin_amdgcn_wmma_f32_16x16x32_bf16(false, af1.b, false, bf1.b, (short)0, acc11, false, false);
  }

  // write out, NCHW mapping; whole block is inside one batch image
  const int bb     = m0 / HWDIM;               // constant per block
  const int hwbase = m0 - bb * HWDIM;
  #pragma unroll
  for (int sm = 0; sm < 2; ++sm) {
    #pragma unroll
    for (int sn = 0; sn < 2; ++sn) {
      v8f acc = (sm == 0) ? (sn == 0 ? acc00 : acc01)
                          : (sn == 0 ? acc10 : acc11);
      int ncol = n0 + wn * 32 + sn * 16 + mr;
      float* Crow = Cb + ((long)bb * Cout + ncol) * HWDIM + hwbase
                  + wm * 32 + sm * 16 + hi * 8;
      #pragma unroll
      for (int r = 0; r < 8; ++r) Crow[r] = acc[r];
    }
  }
}

// ---------------------------------------------------------------------------
// Row L2 norms for q and k: inv = 1 / max(sqrt(sum x^2 over HW), 1e-12)
// blockIdx.x in [0,768): [0,384)=Q rows, [384,768)=K rows (KV ch 0..191)
// ---------------------------------------------------------------------------
__global__ __launch_bounds__(256) void rownorm_kernel(
    const float* __restrict__ Q, const float* __restrict__ KV,
    float* __restrict__ invq, float* __restrict__ invk) {
  __shared__ float red[256];
  int row = blockIdx.x;
  int which = row / 384;
  int bc = row % 384;                      // b*192 + c
  const float* p = (which == 0)
      ? (Q + (long)bc * HWDIM)
      : (KV + ((long)(bc / CDIM) * 384 + (bc % CDIM)) * HWDIM);
  float s = 0.f;
  for (int i = threadIdx.x; i < HWDIM; i += 256) { float v = p[i]; s += v * v; }
  red[threadIdx.x] = s;
  __syncthreads();
  for (int o = 128; o > 0; o >>= 1) {
    if (threadIdx.x < o) red[threadIdx.x] += red[threadIdx.x + o];
    __syncthreads();
  }
  if (threadIdx.x == 0) {
    float inv = 1.0f / fmaxf(sqrtf(red[0]), 1e-12f);
    if (which == 0) invq[bc] = inv; else invk[bc] = inv;
  }
}

// ---------------------------------------------------------------------------
// Per (b,head): S = (qn)(kn)^T over K=9216 via WMMA (2x2 tiles of 16x16),
// *temperature, softmax rows, write attn[12][32][32].
// 128 threads = 4 waves, one 16x16 tile each.
// ---------------------------------------------------------------------------
__global__ __launch_bounds__(128) void attn_qk_kernel(
    const float* __restrict__ Q, const float* __restrict__ KV,
    const float* __restrict__ invq, const float* __restrict__ invk,
    const float* __restrict__ temp, float* __restrict__ attn) {
  __shared__ __align__(16) unsigned short Qs[32][40];
  __shared__ __align__(16) unsigned short Ks[32][40];
  __shared__ float S[32][33];

  const int bh = blockIdx.x;
  const int b = bh / NHEAD, hD = bh % NHEAD;
  const int tid = threadIdx.x, lane = tid & 31, wave = tid >> 5;
  const int tm = wave & 1, tn = wave >> 1;
  const int mr = lane & 15, hi = lane >> 4;

  const float* Qb = Q  + ((long)b * CDIM + hD * HD) * HWDIM;
  const float* Kb = KV + ((long)b * 384  + hD * HD) * HWDIM;
  const int row = tid >> 2;            // 0..31
  const int hwc = (tid & 3) << 3;      // 0,8,16,24
  const float sq = invq[b * CDIM + hD * HD + row];
  const float sk = invk[b * CDIM + hD * HD + row];

  v8f acc = {0.f, 0.f, 0.f, 0.f, 0.f, 0.f, 0.f, 0.f};

  for (int k0 = 0; k0 < HWDIM; k0 += 32) {
    __syncthreads();
    const float* qp = Qb + (long)row * HWDIM + k0 + hwc;
    const float* kp = Kb + (long)row * HWDIM + k0 + hwc;
    #pragma unroll
    for (int i = 0; i < 8; ++i) {
      Qs[row][hwc + i] = f32_to_bf16(qp[i] * sq);
      Ks[row][hwc + i] = f32_to_bf16(kp[i] * sk);
    }
    __syncthreads();

    union { v16u u; v16bf b; } af, bfm;
    {
      int r = tm * 16 + mr;
      u16x8 a0 = *(const u16x8*)(&Qs[r][hi * 8]);
      u16x8 a1 = *(const u16x8*)(&Qs[r][16 + hi * 8]);
      #pragma unroll
      for (int i = 0; i < 8; ++i) { af.u[i] = a0[i]; af.u[8 + i] = a1[i]; }
    }
    {
      int c = tn * 16 + mr;
      u16x8 b0 = *(const u16x8*)(&Ks[c][hi * 8]);
      u16x8 b1 = *(const u16x8*)(&Ks[c][16 + hi * 8]);
      #pragma unroll
      for (int i = 0; i < 8; ++i) { bfm.u[i] = b0[i]; bfm.u[8 + i] = b1[i]; }
    }
    acc = __builtin_amdgcn_wmma_f32_16x16x32_bf16(
        false, af.b, false, bfm.b, (short)0, acc, false, false);
  }

  const float t = temp[hD];
  const int col = tn * 16 + mr;
  #pragma unroll
  for (int r = 0; r < 8; ++r)
    S[tm * 16 + hi * 8 + r][col] = acc[r] * t;
  __syncthreads();

  if (tid < 32) {                      // softmax over the 32 columns of row tid
    float mx = -3.4e38f;
    for (int d = 0; d < 32; ++d) mx = fmaxf(mx, S[tid][d]);
    float sum = 0.f;
    for (int d = 0; d < 32; ++d) { float e = expf(S[tid][d] - mx); S[tid][d] = e; sum += e; }
    float inv = 1.0f / sum;
    for (int d = 0; d < 32; ++d)
      attn[(long)bh * 1024 + tid * 32 + d] = S[tid][d] * inv;
  }
}

// ---------------------------------------------------------------------------
// out[b, head*32+c, hw] = sum_d attn[c][d] * V[d][hw]  (K=32, bf16 output)
// grid: (HW/256, NHEAD, BATCH)
// ---------------------------------------------------------------------------
__global__ __launch_bounds__(256) void attn_av_kernel(
    const float* __restrict__ attn, const float* __restrict__ KV,
    unsigned short* __restrict__ outbf) {
  __shared__ float S[32][33];
  const int b = blockIdx.z, hD = blockIdx.y;
  const int hw = blockIdx.x * 256 + threadIdx.x;
  const int bh = b * NHEAD + hD;
  for (int i = threadIdx.x; i < 1024; i += 256)
    S[i >> 5][i & 31] = attn[(long)bh * 1024 + i];
  __syncthreads();

  const float* Vb = KV + ((long)b * 384 + CDIM + hD * HD) * HWDIM;
  float acc[32];
  #pragma unroll
  for (int c = 0; c < 32; ++c) acc[c] = 0.f;
  for (int d = 0; d < 32; ++d) {
    float v = Vb[(long)d * HWDIM + hw];
    #pragma unroll
    for (int c = 0; c < 32; ++c) acc[c] += S[c][d] * v;
  }
  unsigned short* ob = outbf + ((long)b * CDIM + hD * HD) * HWDIM + hw;
  #pragma unroll
  for (int c = 0; c < 32; ++c) ob[(long)c * HWDIM] = f32_to_bf16(acc[c]);
}

// ---------------------------------------------------------------------------
// Host-side orchestration
// ---------------------------------------------------------------------------
extern "C" void kernel_launch(void* const* d_in, const int* in_sizes, int n_in,
                              void* d_out, int out_size, void* d_ws, size_t ws_size,
                              hipStream_t stream) {
  const float* x         = (const float*)d_in[0];
  const float* p_conv_w  = (const float*)d_in[1];
  const float* p_conv_b  = (const float*)d_in[2];
  const float* dconv_w   = (const float*)d_in[3];
  const float* qkv_w     = (const float*)d_in[4];
  const float* proj_w    = (const float*)d_in[5];
  const float* temp      = (const float*)d_in[6];

  // workspace carve-up (256B aligned)
  char* base = (char*)d_ws;
  size_t off = 0;
  auto carve = [&](size_t bytes) {
    void* p = base + off;
    off = (off + bytes + 255) & ~(size_t)255;
    return p;
  };
  unsigned short* Abuf = (unsigned short*)carve((size_t)KDEF * MTOT * 2);          // 63.7 MB (im2col, then deform samples)
  float*          OFF  = (float*)carve((size_t)BATCH * NOFF * HWDIM * 4);          // 4.7 MB
  float*          Qb   = (float*)carve((size_t)BATCH * CDIM * HWDIM * 4);          // 14.2 MB
  float*          KVb  = (float*)carve((size_t)BATCH * 384 * HWDIM * 4);           // 28.3 MB
  unsigned short* XBF  = (unsigned short*)carve((size_t)BATCH * CDIM * HWDIM * 2); // 7.1 MB
  unsigned short* AOUT = (unsigned short*)carve((size_t)BATCH * CDIM * HWDIM * 2); // 7.1 MB
  unsigned short* Wd   = (unsigned short*)carve((size_t)KDEF * CDIM * 2);
  unsigned short* Wo   = (unsigned short*)carve((size_t)KDEF * NOFF * 2);
  unsigned short* Wkv  = (unsigned short*)carve((size_t)CDIM * 384 * 2);
  unsigned short* Wp   = (unsigned short*)carve((size_t)CDIM * CDIM * 2);
  float*          ATTN = (float*)carve((size_t)BATCH * NHEAD * 1024 * 4);
  float*          INVQ = (float*)carve((size_t)BATCH * CDIM * 4);
  float*          INVK = (float*)carve((size_t)BATCH * CDIM * 4);
  (void)ws_size; (void)n_in; (void)in_sizes; (void)out_size;

  // ---- weight repacks (B matrices, row-major [K][N], bf16) ----
  pack_transpose_pad<<<(KDEF * CDIM + 255) / 256, 256, 0, stream>>>(dconv_w, Wd, CDIM, KDEF, CDIM);
  pack_transpose_pad<<<(KDEF * NOFF + 255) / 256, 256, 0, stream>>>(p_conv_w, Wo, 18, KDEF, NOFF);
  pack_transpose_pad<<<(CDIM * 384 + 255) / 256, 256, 0, stream>>>(qkv_w + CDIM * CDIM, Wkv, 384, CDIM, 384);
  pack_transpose_pad<<<(CDIM * CDIM + 255) / 256, 256, 0, stream>>>(proj_w, Wp, CDIM, CDIM, CDIM);

  // ---- x -> bf16 (A^T layout for the KV GEMM: [c][hw] per batch) ----
  f32_to_bf16_copy<<<(int)(((long)BATCH * CDIM * HWDIM + 255) / 256), 256, 0, stream>>>(
      x, XBF, (long)BATCH * CDIM * HWDIM);

  // ---- offset conv: im2col + GEMM (N padded 18 -> 64) ----
  im2col_kernel<<<(int)(((long)KDEF * MTOT + 255) / 256), 256, 0, stream>>>(x, Abuf);
  gemm_bf16_nchw<<<dim3(NOFF / 64, MTOT / 128, 1), 256, 0, stream>>>(
      Abuf, Wo, OFF, KDEF, MTOT, NOFF, 0, 0, NOFF);

  // ---- deformable sampling (overwrites Abuf) + deform GEMM -> Q ----
  deform_sample_kernel<<<(BATCH * 9 * HWDIM + 255) / 256, 256, 0, stream>>>(
      x, OFF, p_conv_b, Abuf);
  gemm_bf16_nchw<<<dim3(CDIM / 64, MTOT / 128, 1), 256, 0, stream>>>(
      Abuf, Wd, Qb, KDEF, MTOT, CDIM, 0, 0, CDIM);

  // ---- KV projection (batched over B) ----
  gemm_bf16_nchw<<<dim3(384 / 64, HWDIM / 128, BATCH), 256, 0, stream>>>(
      XBF, Wkv, KVb, CDIM, HWDIM, 384,
      (long)CDIM * HWDIM, (long)384 * HWDIM, 384);

  // ---- attention ----
  rownorm_kernel<<<768, 256, 0, stream>>>(Qb, KVb, INVQ, INVK);
  attn_qk_kernel<<<BATCH * NHEAD, 128, 0, stream>>>(Qb, KVb, INVQ, INVK, temp, ATTN);
  attn_av_kernel<<<dim3(HWDIM / 256, NHEAD, BATCH), 256, 0, stream>>>(ATTN, KVb, AOUT);

  // ---- final projection -> d_out (f32, NCHW) ----
  gemm_bf16_nchw<<<dim3(CDIM / 64, HWDIM / 128, BATCH), 256, 0, stream>>>(
      AOUT, Wp, (float*)d_out, CDIM, HWDIM, CDIM,
      (long)CDIM * HWDIM, (long)CDIM * HWDIM, CDIM);
}
